// GRUD_87522843558392
// MI455X (gfx1250) — matrix operand order repaired and checked by reading, not compiled, not run on hardware
//
#include <hip/hip_runtime.h>
#include <hip/hip_bf16.h>
#include <math.h>

// ---- shapes ----
#define BB 128
#define DD 256
#define HH 1024
#define TT 512
#define KCAT 1280   // D + H
#define N3 3072     // 3*H
#define NBLK 96     // persistent workgroups (= stage-1 tile count)

typedef _Float16 f16;
typedef _Float16 v8h  __attribute__((ext_vector_type(8)));
typedef _Float16 v16h __attribute__((ext_vector_type(16)));
typedef float    v8f  __attribute__((ext_vector_type(8)));

// Per-lane async copy: 16 bytes global -> LDS, tracked by ASYNCcnt.
// lds: raw LDS byte offset (low 32 bits of the flat shared address),
// gp : per-lane 64-bit global address.
#define ASYNC_LD16(lds, gp)                                              \
    asm volatile("global_load_async_to_lds_b128 %0, %1, off"             \
                 :: "v"(lds), "v"(gp) : "memory")

static __device__ __forceinline__ unsigned lds_off(const void* p) {
    return (unsigned)(uintptr_t)p;      // flat LDS addr: offset in [31:0]
}

static __device__ __forceinline__ float sigf(float x) {
    return 1.0f / (1.0f + expf(-x));
}

static __device__ __forceinline__ v16h cat8(v8h lo, v8h hi) {
    return __builtin_shufflevector(lo, hi, 0,1,2,3,4,5,6,7,8,9,10,11,12,13,14,15);
}

// Device-wide sense/generation barrier. All NBLK blocks are resident (96
// blocks x 4 waves x 20.5KB LDS on a CDNA5 part -> co-residency guaranteed).
// Release on arrival / acquire on the spin at AGENT scope makes the plain
// stores to the f16/f32 state buffers visible across WGPs.
static __device__ __forceinline__ void grid_barrier(unsigned* cnt, unsigned* gen)
{
    __syncthreads();
    if (threadIdx.x == 0) {
        unsigned g = __hip_atomic_load(gen, __ATOMIC_RELAXED,
                                       __HIP_MEMORY_SCOPE_AGENT);
        unsigned a = __hip_atomic_fetch_add(cnt, 1u, __ATOMIC_ACQ_REL,
                                            __HIP_MEMORY_SCOPE_AGENT);
        if (a == (unsigned)(NBLK - 1)) {
            __hip_atomic_store(cnt, 0u, __ATOMIC_RELAXED,
                               __HIP_MEMORY_SCOPE_AGENT);
            __hip_atomic_fetch_add(gen, 1u, __ATOMIC_RELEASE,
                                   __HIP_MEMORY_SCOPE_AGENT);
        } else {
            while (__hip_atomic_load(gen, __ATOMIC_ACQUIRE,
                                     __HIP_MEMORY_SCOPE_AGENT) == g) {
                __builtin_amdgcn_s_sleep(2);
            }
        }
    }
    __syncthreads();
}

// ---------------------------------------------------------------------------
// Pack weights (fp32, row-major [K][H]) into f16 transposed W_t[n][k] so both
// GEMM operands load contiguous-K (feeds the 16x32 / 32x16 fragment layouts).
// ---------------------------------------------------------------------------
__global__ __launch_bounds__(256) void k_pack_w(
    const float* __restrict__ w_xz, const float* __restrict__ w_hz,
    const float* __restrict__ w_xr, const float* __restrict__ w_hr,
    const float* __restrict__ w_xh, const float* __restrict__ w_hh,
    f16* __restrict__ Wt)
{
    int idx = blockIdx.x * 256 + threadIdx.x;           // < 3072*1280
    int n = idx / KCAT;
    int k = idx - n * KCAT;
    int g = n >> 10;
    int c = n & (HH - 1);
    float v;
    if (k < DD) {
        const float* wx = (g == 0) ? w_xz : (g == 1) ? w_xr : w_xh;
        v = wx[k * HH + c];
    } else {
        const float* wh = (g == 0) ? w_hz : (g == 1) ? w_hr : w_hh;
        v = wh[(k - DD) * HH + c];
    }
    Wt[idx] = (f16)v;
}

// ---------------------------------------------------------------------------
// Transpose value[B][D][T] f32 -> xall[T][B][D] f16 (LDS-tiled transpose)
// ---------------------------------------------------------------------------
__global__ __launch_bounds__(256) void k_pack_x(
    const float* __restrict__ value, f16* __restrict__ xall)
{
    __shared__ float tile[32][33];
    int t0 = blockIdx.x << 5;
    int d0 = blockIdx.y << 5;
    int b  = blockIdx.z;
    int tx = threadIdx.x & 31;
    int ty = threadIdx.x >> 5;
#pragma unroll
    for (int i = 0; i < 4; ++i) {
        int dl = ty + (i << 3);
        tile[dl][tx] = value[((size_t)b * DD + d0 + dl) * TT + t0 + tx];
    }
    __syncthreads();
#pragma unroll
    for (int i = 0; i < 4; ++i) {
        int tl = ty + (i << 3);
        xall[((size_t)(t0 + tl) * BB + b) * DD + d0 + tx] = (f16)tile[tx][tl];
    }
}

// ---------------------------------------------------------------------------
// Zero hidden state + barrier counters (every call -> replay-deterministic).
// ---------------------------------------------------------------------------
__global__ __launch_bounds__(256) void k_init(
    float* __restrict__ h32, f16* __restrict__ h16,
    unsigned* __restrict__ cnt, unsigned* __restrict__ gen)
{
    int idx = blockIdx.x * 256 + threadIdx.x;           // < B*H
    h32[idx] = 0.0f;
    h16[idx] = (f16)0.0f;
    if (idx == 0) { cnt[0] = 0u; gen[0] = 0u; }
}

// ---------------------------------------------------------------------------
// Persistent scan kernel: all 512 steps, 2 device-wide barriers per step.
// Tiles are staged with double-buffered async-to-LDS copies (ASYNCcnt):
// tile k+1 is in flight while tile k is consumed by WMMA.
//
// Phase A (96 blocks, tile 64x64, 4 waves @ 32x32):
//   [x_t ; h] (K=1280; x-only K=256 for gate-h cols) @ W_t
//   -> z32 = sigmoid(+b_z), rh16 = f16(sigmoid(+b_r)*h), xh32 = +b_h
// Phase B (64 blocks, tile 32x64, 4 waves @ 16x32):
//   (r*h) @ W_hh; h~ = tanh(xh32 + .); h' = (1-z)h + z h~
//   -> out[b,t,:]; h32/h16 = gamma_{t+1}*h' (or hlast at t=T-1)
// ---------------------------------------------------------------------------
__global__ __launch_bounds__(128) void k_scan(
    const f16*  __restrict__ xall,
    const f16*  __restrict__ Wt,
    const float* __restrict__ b_z,
    const float* __restrict__ b_r,
    const float* __restrict__ b_h,
    float* __restrict__ h32,
    f16*   __restrict__ h16,
    float* __restrict__ z32,
    f16*   __restrict__ rh16,
    float* __restrict__ xh32,
    const float* __restrict__ delta,
    const float* __restrict__ w_dg,
    const float* __restrict__ b_dg,
    float* __restrict__ out,      // [B,T,H]
    float* __restrict__ hlast,    // [B,H]
    unsigned* cnt, unsigned* gen)
{
    __shared__ __align__(16) f16 As[2][64][40];
    __shared__ __align__(16) f16 Bs[2][64][40];

    const int blk  = blockIdx.x;
    const int tid  = threadIdx.x;
    const int lane = tid & 31;
    const int wid  = tid >> 5;
    const int fm   = lane & 15;
    const int nl   = lane & 15;
    const int kA   = (lane >> 4) << 3;      // A frag interleave base 0/8
    const int kB   = (lane >> 4) << 3;      // B frag interleave base 0/8
    const int mh   = (lane >> 4) << 3;      // C frag row offset 0/8

    const int wm   = wid >> 1;              // 0..1
    const int wn   = wid & 1;               // 0..1
    const int lrow = tid >> 1;              // 0..63 (64-row tile loads)
    const int lk   = (tid & 1) << 4;        // 0/16
    const int arow = tid >> 2;              // 0..31 (32-row tile loads)
    const int ak   = (tid & 3) << 3;        // 0/8/16/24

    // phase-A tile
    const int s1m0 = (blk & 1) << 6;        // 0/64
    const int s1c0 = (blk >> 1) << 6;       // 0..3008
    const int s1K  = (s1c0 >= 2 * HH) ? DD : KCAT;
    // phase-B tile (blocks 0..63)
    const int s2m0 = (blk & 3) << 5;        // 0..96
    const int s2c0 = (blk >> 2) << 6;       // 0..960

    // LDS destinations (raw LDS byte offsets) for async copies
    const unsigned aL[2] = { lds_off(&As[0][lrow][lk]), lds_off(&As[1][lrow][lk]) };
    const unsigned bL[2] = { lds_off(&Bs[0][lrow][lk]), lds_off(&Bs[1][lrow][lk]) };
    const unsigned aL2[2] = { lds_off(&As[0][arow][ak]), lds_off(&As[1][arow][ak]) };

    // issue one phase-A k-tile (4 x 16B per thread) into buffer `sel`
    auto issueA = [&](int t, int kk, int sel) {
        const f16* ap = (kk < DD)
            ? xall + (((size_t)t * BB + (s1m0 + lrow)) << 8) + kk + lk
            : h16  + ((size_t)(s1m0 + lrow) << 10) + (kk - DD) + lk;
        const f16* bp = Wt + (size_t)(s1c0 + lrow) * KCAT + kk + lk;
        ASYNC_LD16(aL[sel],      ap);
        ASYNC_LD16(aL[sel] + 16, ap + 8);
        ASYNC_LD16(bL[sel],      bp);
        ASYNC_LD16(bL[sel] + 16, bp + 8);
    };
    // issue one phase-B k-tile (3 x 16B per thread) into buffer `sel`
    auto issueB = [&](int kk, int sel) {
        const f16* ap = rh16 + ((size_t)(s2m0 + arow) << 10) + kk + ak;
        const f16* bp = Wt + (size_t)(2 * HH + s2c0 + lrow) * KCAT + DD + kk + lk;
        ASYNC_LD16(aL2[sel],     ap);
        ASYNC_LD16(bL[sel],      bp);
        ASYNC_LD16(bL[sel] + 16, bp + 8);
    };

#pragma unroll 1
    for (int t = 0; t < TT; ++t) {
        // ================= phase A =================
        {
            v8f acc[2][2];
#pragma unroll
            for (int i = 0; i < 2; ++i)
#pragma unroll
                for (int j = 0; j < 2; ++j)
                    acc[i][j] = (v8f){0.f,0.f,0.f,0.f,0.f,0.f,0.f,0.f};

            issueA(t, 0, 0);
            int sel = 0;
#pragma unroll 1
            for (int kk = 0; kk < s1K; kk += 32) {
                __syncthreads();            // buffer sel^1 fully consumed
                if (kk + 32 < s1K) {
                    issueA(t, kk + 32, sel ^ 1);
                    asm volatile("s_wait_asynccnt 0x4" ::: "memory");
                } else {
                    asm volatile("s_wait_asynccnt 0x0" ::: "memory");
                }
                __syncthreads();            // all waves' tile-k data landed

                v16h af[2], bf[2];
#pragma unroll
                for (int sm = 0; sm < 2; ++sm) {
                    int r = (wm << 5) + (sm << 4) + fm;
                    af[sm] = cat8(*(const v8h*)&As[sel][r][kA],
                                  *(const v8h*)&As[sel][r][kA + 16]);
                }
#pragma unroll
                for (int sn = 0; sn < 2; ++sn) {
                    int c = (wn << 5) + (sn << 4) + fm;
                    bf[sn] = cat8(*(const v8h*)&Bs[sel][c][kB],
                                  *(const v8h*)&Bs[sel][c][kB + 16]);
                }
#pragma unroll
                for (int sm = 0; sm < 2; ++sm)
#pragma unroll
                    for (int sn = 0; sn < 2; ++sn)
                        acc[sm][sn] = __builtin_amdgcn_wmma_f32_16x16x32_f16(
                            false, af[sm], false, bf[sn], (short)0,
                            acc[sm][sn], false, false);
                sel ^= 1;
            }

            const int mbase = s1m0 + (wm << 5) + mh;
            const int nbase = s1c0 + (wn << 5);
#pragma unroll
            for (int sm = 0; sm < 2; ++sm)
#pragma unroll
                for (int sn = 0; sn < 2; ++sn)
#pragma unroll
                    for (int g = 0; g < 8; ++g) {
                        int m = mbase + (sm << 4) + g;
                        int n = nbase + (sn << 4) + nl;
                        float v = acc[sm][sn][g];
                        if (n < HH) {
                            z32[(m << 10) + n] = sigf(v + b_z[n]);
                        } else if (n < 2 * HH) {
                            int c = n - HH;
                            float r = sigf(v + b_r[c]);
                            rh16[(m << 10) + c] =
                                (f16)(r * h32[(m << 10) + c]);
                        } else {
                            int c = n - 2 * HH;
                            xh32[(m << 10) + c] = v + b_h[c];
                        }
                    }
        }
        grid_barrier(cnt, gen);

        // ================= phase B =================
        if (blk < 64) {
            v8f acc[2];
#pragma unroll
            for (int j = 0; j < 2; ++j)
                acc[j] = (v8f){0.f,0.f,0.f,0.f,0.f,0.f,0.f,0.f};

            issueB(0, 0);
            int sel = 0;
#pragma unroll 1
            for (int kk = 0; kk < HH; kk += 32) {
                __syncthreads();
                if (kk + 32 < HH) {
                    issueB(kk + 32, sel ^ 1);
                    asm volatile("s_wait_asynccnt 0x3" ::: "memory");
                } else {
                    asm volatile("s_wait_asynccnt 0x0" ::: "memory");
                }
                __syncthreads();

                v16h af, bf[2];
                {
                    int r = (wm << 4) + fm;
                    af = cat8(*(const v8h*)&As[sel][r][kA],
                              *(const v8h*)&As[sel][r][kA + 16]);
                }
#pragma unroll
                for (int sn = 0; sn < 2; ++sn) {
                    int c = (wn << 5) + (sn << 4) + fm;
                    bf[sn] = cat8(*(const v8h*)&Bs[sel][c][kB],
                                  *(const v8h*)&Bs[sel][c][kB + 16]);
                }
#pragma unroll
                for (int sn = 0; sn < 2; ++sn)
                    acc[sn] = __builtin_amdgcn_wmma_f32_16x16x32_f16(
                        false, af, false, bf[sn], (short)0,
                        acc[sn], false, false);
                sel ^= 1;
            }

            const float dnext = (t + 1 < TT) ? delta[t + 1] : 0.0f;
            const int mbase = s2m0 + (wm << 4) + mh;
            const int nbase = s2c0 + (wn << 5);
#pragma unroll
            for (int sn = 0; sn < 2; ++sn)
#pragma unroll
                for (int g = 0; g < 8; ++g) {
                    int m = mbase + g;
                    int c = nbase + (sn << 4) + nl;
                    int idx = (m << 10) + c;
                    float pre = acc[sn][g] + xh32[idx];
                    float ht  = tanhf(pre);
                    float z   = z32[idx];
                    float hp  = h32[idx];
                    float hn  = hp + z * (ht - hp);   // (1-z)h + z*h~
                    out[((size_t)m * TT + t) * HH + c] = hn;
                    if (t == TT - 1) {
                        hlast[idx] = hn;
                    } else {
                        float gm = expf(-fmaxf(0.0f,
                                    fmaf(w_dg[c], dnext, b_dg[c])));
                        float hs = gm * hn;
                        h32[idx] = hs;
                        h16[idx] = (f16)hs;
                    }
                }
        }
        grid_barrier(cnt, gen);
    }
}

// ---------------------------------------------------------------------------
extern "C" void kernel_launch(void* const* d_in, const int* in_sizes, int n_in,
                              void* d_out, int out_size, void* d_ws, size_t ws_size,
                              hipStream_t stream)
{
    const float* value = (const float*)d_in[0];
    const float* delta = (const float*)d_in[1];
    const float* w_dg  = (const float*)d_in[2];
    const float* w_xz  = (const float*)d_in[3];
    const float* w_hz  = (const float*)d_in[4];
    const float* w_xr  = (const float*)d_in[5];
    const float* w_hr  = (const float*)d_in[6];
    const float* w_xh  = (const float*)d_in[7];
    const float* w_hh  = (const float*)d_in[8];
    const float* b_dg  = (const float*)d_in[9];
    const float* b_z   = (const float*)d_in[10];
    const float* b_r   = (const float*)d_in[11];
    const float* b_h   = (const float*)d_in[12];

    char* ws = (char*)d_ws;
    size_t off = 0;
    f16*   Wt   = (f16*)(ws + off);   off += (size_t)N3 * KCAT * sizeof(f16);
    f16*   xall = (f16*)(ws + off);   off += (size_t)TT * BB * DD * sizeof(f16);
    float* h32  = (float*)(ws + off); off += (size_t)BB * HH * sizeof(float);
    f16*   h16  = (f16*)(ws + off);   off += (size_t)BB * HH * sizeof(f16);
    float* z32  = (float*)(ws + off); off += (size_t)BB * HH * sizeof(float);
    f16*   rh16 = (f16*)(ws + off);   off += (size_t)BB * HH * sizeof(f16);
    float* xh32 = (float*)(ws + off); off += (size_t)BB * HH * sizeof(float);
    unsigned* cnt = (unsigned*)(ws + off); off += 256;
    unsigned* gen = (unsigned*)(ws + off); off += 256;

    float* out   = (float*)d_out;                       // [B,T,H]
    float* hlast = out + (size_t)BB * TT * HH;          // [B,H]

    k_pack_w<<<(N3 * KCAT) / 256, 256, 0, stream>>>(w_xz, w_hz, w_xr, w_hr,
                                                    w_xh, w_hh, Wt);
    k_pack_x<<<dim3(TT / 32, DD / 32, BB), 256, 0, stream>>>(value, xall);
    k_init<<<(BB * HH) / 256, 256, 0, stream>>>(h32, h16, cnt, gen);

    // one persistent kernel runs the whole 512-step scan
    k_scan<<<NBLK, 128, 0, stream>>>(xall, Wt, b_z, b_r, b_h,
                                     h32, h16, z32, rh16, xh32,
                                     delta, w_dg, b_dg, out, hlast, cnt, gen);
}
